// GraphConvModel_67774583931092
// MI455X (gfx1250) — matrix operand order repaired and checked by reading, not compiled, not run on hardware
//
#include <hip/hip_runtime.h>

typedef __attribute__((ext_vector_type(16))) _Float16 v16h;
typedef __attribute__((ext_vector_type(8)))  float    v8f;

#define NN   50000
#define EE   800000
#define HH   256
#define KT   512      // combined K: [W_rel ; W_root]
#define GG   64
#define OUTD 24
#define NTILE (NN / 16)        // 3125 row tiles
#define TILEH 8192             // halfs per A tile: 16 ksteps * 32 lanes * 16 halfs

union V16U { v16h h; float4 f4[2]; };

// ---------------- utility: zero float4 ----------------
__global__ void k_zero4(float4* p, int n4) {
    int t = blockIdx.x * blockDim.x + threadIdx.x;
    if (t < n4) p[t] = make_float4(0.f, 0.f, 0.f, 0.f);
}

// ------------- pack [W_rel;W_root] -> f16, transposed (c-major, k contiguous) -------------
// wt[layer][c][k]  (k<256: w_rel[k][c], else w_root[k-256][c])
__global__ void k_pack_w(const float* __restrict__ w_rel, const float* __restrict__ w_root,
                         _Float16* __restrict__ wt) {
    int idx = blockIdx.x * blockDim.x + threadIdx.x;
    if (idx >= 6 * HH * KT) return;
    int layer = idx / (HH * KT);
    int rem   = idx % (HH * KT);
    int c = rem / KT;
    int k = rem % KT;
    float v = (k < HH) ? w_rel [layer * HH * HH + k * HH + c]
                       : w_root[layer * HH * HH + (k - HH) * HH + c];
    wt[idx] = (_Float16)v;
}

// ---------------- layer 1: scatter of x (F_IN=3) ----------------
__global__ void k_scatter3(const float* __restrict__ x, const int* __restrict__ ei,
                           float* __restrict__ agg3) {
    int e = blockIdx.x * blockDim.x + threadIdx.x;
    if (e >= EE) return;
    int s = ei[e], d = ei[EE + e];
    atomicAdd(&agg3[d * 3 + 0], x[s * 3 + 0]);
    atomicAdd(&agg3[d * 3 + 1], x[s * 3 + 1]);
    atomicAdd(&agg3[d * 3 + 2], x[s * 3 + 2]);
}

// layer 1 dense: h = relu(agg3 @ w_rel1 + x @ w_root1 + b1), F_IN=3 -> scalar is cheap
__global__ void k_layer1(const float* __restrict__ x, const float* __restrict__ agg3,
                         const float* __restrict__ wr, const float* __restrict__ wro,
                         const float* __restrict__ b1, float* __restrict__ h) {
    int t = blockIdx.x * blockDim.x + threadIdx.x;   // N*H threads
    int n = t >> 8, c = t & (HH - 1);
    float acc = b1[c];
#pragma unroll
    for (int k = 0; k < 3; ++k)
        acc += agg3[n * 3 + k] * wr[k * HH + c] + x[n * 3 + k] * wro[k * HH + c];
    h[t] = acc > 0.f ? acc : 0.f;
}

// ---------------- H-wide edge scatter: agg[dst] += h[src] ----------------
// thread = (edge, 32-channel chunk): float4 loads + f32 atomics (L2-resident)
__global__ void k_scatter(const int* __restrict__ ei, const float* __restrict__ h,
                          float* __restrict__ agg) {
    int t = blockIdx.x * blockDim.x + threadIdx.x;   // E*8 threads
    int e = t >> 3, part = t & 7;
    int s = ei[e], d = ei[EE + e];
    const float4* hp = (const float4*)(h + (size_t)s * HH + part * 32);
    float* ap = agg + (size_t)d * HH + part * 32;
#pragma unroll
    for (int j = 0; j < 8; ++j) {
        float4 v = hp[j];
        atomicAdd(ap + j * 4 + 0, v.x);
        atomicAdd(ap + j * 4 + 1, v.y);
        atomicAdd(ap + j * 4 + 2, v.z);
        atomicAdd(ap + j * 4 + 3, v.w);
    }
}

// ---------------- pack activations into WMMA-fragment-ordered f16 A buffer ----------------
// A16 layout: [rowtile rb][kstep ks][lane][16 halfs] with the 16-bit A 16x32 VGPR layout:
//   a[0..7]  = A[row][k0 + hs*8 + j],  a[8..15] = A[row][k0 + 16 + hs*8 + j]
//   row = rb*16 + lane%16, hs = lane/16, combined k: [agg | hin]
__global__ void k_packA(const float* __restrict__ agg, const float* __restrict__ hin,
                        _Float16* __restrict__ A16) {
    int t = blockIdx.x * blockDim.x + threadIdx.x;   // NTILE*16*32 = 1.6M threads
    int lane = t & 31;
    int ks   = (t >> 5) & 15;
    int rb   = t >> 9;
    int l16 = lane & 15, hs = lane >> 4;
    int row = (rb << 4) + l16;
    int k0  = ks << 5;                // combined-K base (0..480)
    int kA  = k0 + (hs << 3);         // first 8-wide chunk
    int kB  = kA + 16;                // second 8-wide chunk (never straddles 256)
    const float* pA = (kA < HH) ? (agg + (size_t)row * HH + kA)
                                : (hin + (size_t)row * HH + (kA - HH));
    const float* pB = (kB < HH) ? (agg + (size_t)row * HH + kB)
                                : (hin + (size_t)row * HH + (kB - HH));
    float4 A0 = ((const float4*)pA)[0];
    float4 A1 = ((const float4*)pA)[1];
    float4 A2 = ((const float4*)pB)[0];
    float4 A3 = ((const float4*)pB)[1];
    v16h a;
    a[0]=(_Float16)A0.x;  a[1]=(_Float16)A0.y;  a[2]=(_Float16)A0.z;  a[3]=(_Float16)A0.w;
    a[4]=(_Float16)A1.x;  a[5]=(_Float16)A1.y;  a[6]=(_Float16)A1.z;  a[7]=(_Float16)A1.w;
    a[8]=(_Float16)A2.x;  a[9]=(_Float16)A2.y;  a[10]=(_Float16)A2.z; a[11]=(_Float16)A2.w;
    a[12]=(_Float16)A3.x; a[13]=(_Float16)A3.y; a[14]=(_Float16)A3.z; a[15]=(_Float16)A3.w;
    *(v16h*)(A16 + (size_t)t * 16) = a;
}

// ---------------- fused dual-GEMM + bias + ReLU via WMMA f16->f32 ----------------
// hout = relu( [agg|hin] (N x 512 f16, pre-swizzled) @ wt^T (512 x 256 f16) + bias )
// block = 256 thr = 8 waves; block M-tile = 16 rows staged in LDS; wave owns 2 N-tiles.
__global__ void __launch_bounds__(256)
k_gemm(const _Float16* __restrict__ A16, const _Float16* __restrict__ wt,
       const float* __restrict__ bias, float* __restrict__ hout) {
    __shared__ __align__(16) _Float16 Atile[TILEH];   // 16 KB

    const int tid  = threadIdx.x;
    const int lane = tid & 31;
    const int wave = tid >> 5;
    const int l16  = lane & 15;
    const int hs   = lane >> 4;
    const int rowBase = blockIdx.x << 4;

    // stage this block's 16KB A tile into LDS (coalesced b128 loads)
    const float4* gsrc = (const float4*)(A16 + (size_t)blockIdx.x * TILEH);
    float4* ldst = (float4*)Atile;
#pragma unroll
    for (int j = 0; j < 4; ++j) ldst[tid + j * 256] = gsrc[tid + j * 256];

    // speculative L2 prefetch of the next block's A tile
    if (blockIdx.x + 1 < gridDim.x)
        __builtin_prefetch(A16 + (size_t)(blockIdx.x + 1) * TILEH, 0, 1);

    __syncthreads();

    const int col1 = (wave << 5) + l16;
    const int col2 = col1 + 16;
    const _Float16* wrow1 = wt + (size_t)col1 * KT;
    const _Float16* wrow2 = wt + (size_t)col2 * KT;

    v8f acc1 = {};
    v8f acc2 = {};
#pragma unroll
    for (int ks = 0; ks < 16; ++ks) {
        // A fragment straight out of LDS (pre-swizzled, no cvt on hot path)
        v16h a = *(const v16h*)(Atile + (((ks << 5) + lane) << 4));
        // B fragments: 32x16 layout, lane = column; b[0..7] = k0+hs*8+j, b[8..15] = k0+16+hs*8+j
        const int kb = (ks << 5) + (hs << 3);
        V16U b1, b2;
        const float4* bp1 = (const float4*)(wrow1 + kb);
        const float4* bp2 = (const float4*)(wrow2 + kb);
        b1.f4[0] = bp1[0]; b1.f4[1] = bp1[2];
        b2.f4[0] = bp2[0]; b2.f4[1] = bp2[2];

        acc1 = __builtin_amdgcn_wmma_f32_16x16x32_f16(false, a, false, b1.h,
                                                      (short)0, acc1, false, false);
        acc2 = __builtin_amdgcn_wmma_f32_16x16x32_f16(false, a, false, b2.h,
                                                      (short)0, acc2, false, false);
    }

    // C/D layout: VGPR r -> M = r + 8*hs, N = l16
    float bv1 = bias[col1], bv2 = bias[col2];
    const int rOff = hs << 3;
#pragma unroll
    for (int r = 0; r < 8; ++r) {
        size_t row = (size_t)(rowBase + rOff + r);
        float v1 = acc1[r] + bv1;
        float v2 = acc2[r] + bv2;
        hout[row * HH + col1] = v1 > 0.f ? v1 : 0.f;
        hout[row * HH + col2] = v2 > 0.f ? v2 : 0.f;
    }
}

// ---------------- global mean pool ----------------
__global__ void k_pool_count(const int* __restrict__ batch, float* __restrict__ counts) {
    int n = blockIdx.x * blockDim.x + threadIdx.x;
    if (n < NN) atomicAdd(&counts[batch[n]], 1.0f);
}

__global__ void k_pool_sum(const int* __restrict__ batch, const float* __restrict__ h,
                           float* __restrict__ pooled) {
    int t = blockIdx.x * blockDim.x + threadIdx.x;   // N * 64 threads (4 channels each)
    int n = t >> 6, q = t & 63;
    int g = batch[n];
    float4 v = ((const float4*)(h + (size_t)n * HH))[q];
    float* p = pooled + g * HH + q * 4;
    atomicAdd(p + 0, v.x);
    atomicAdd(p + 1, v.y);
    atomicAdd(p + 2, v.z);
    atomicAdd(p + 3, v.w);
}

// ---------------- output head: (pooled/cnt) @ w_out + b_out ----------------
__global__ void k_head(const float* __restrict__ pooled, const float* __restrict__ counts,
                       const float* __restrict__ w_out, const float* __restrict__ b_out,
                       float* __restrict__ out) {
    int t = blockIdx.x * blockDim.x + threadIdx.x;
    if (t >= GG * OUTD) return;
    int g = t / OUTD, o = t % OUTD;
    float inv = 1.f / fmaxf(counts[g], 1.f);
    float s = 0.f;
    for (int c = 0; c < HH; ++c) s += pooled[g * HH + c] * w_out[c * OUTD + o];
    out[t] = s * inv + b_out[o];
}

extern "C" void kernel_launch(void* const* d_in, const int* in_sizes, int n_in,
                              void* d_out, int out_size, void* d_ws, size_t ws_size,
                              hipStream_t stream) {
    (void)in_sizes; (void)n_in; (void)out_size; (void)ws_size;

    const float* x       = (const float*)d_in[0];
    const int*   ei      = (const int*)d_in[1];
    const int*   batch   = (const int*)d_in[2];
    const float* w_rel1  = (const float*)d_in[3];
    const float* w_root1 = (const float*)d_in[4];
    const float* b1      = (const float*)d_in[5];
    const float* w_rel   = (const float*)d_in[6];
    const float* w_root  = (const float*)d_in[7];
    const float* bvec    = (const float*)d_in[8];
    const float* w_out   = (const float*)d_in[9];
    const float* b_out   = (const float*)d_in[10];
    float* out = (float*)d_out;

    // workspace carve-out (all state re-initialized every call)
    char* ws = (char*)d_ws;
    size_t off = 0;
    auto carve = [&](size_t bytes) -> void* {
        void* p = ws + off;
        off = (off + bytes + 255) & ~(size_t)255;
        return p;
    };
    float*    agg    = (float*)   carve((size_t)NN * HH * 4);
    float*    hA     = (float*)   carve((size_t)NN * HH * 4);
    float*    hB     = (float*)   carve((size_t)NN * HH * 4);
    _Float16* A16    = (_Float16*)carve((size_t)NTILE * TILEH * 2);   // 51.2 MB
    _Float16* wt16   = (_Float16*)carve((size_t)6 * HH * KT * 2);
    float*    agg3   = (float*)   carve((size_t)NN * 3 * 4);
    float*    pooled = (float*)   carve((size_t)GG * HH * 4);
    float*    counts = (float*)   carve((size_t)GG * 4);

    dim3 B(256);

    // weights -> f16 transposed pack (all 6 layers)
    k_pack_w<<<(6 * HH * KT + 255) / 256, B, 0, stream>>>(w_rel, w_root, wt16);

    // zero small state
    k_zero4<<<(NN * 3 / 4 + 255) / 256, B, 0, stream>>>((float4*)agg3, NN * 3 / 4);
    k_zero4<<<(GG * HH / 4 + 255) / 256, B, 0, stream>>>((float4*)pooled, GG * HH / 4);
    k_zero4<<<1, B, 0, stream>>>((float4*)counts, GG / 4);

    // layer 1 (F_IN=3)
    k_scatter3<<<(EE + 255) / 256, B, 0, stream>>>(x, ei, agg3);
    k_layer1<<<(NN * HH) / 256, B, 0, stream>>>(x, agg3, w_rel1, w_root1, b1, hA);

    // layers 2..7: scatter-add + pack + fused dual GEMM (WMMA) + bias + ReLU
    float* hc = hA;
    float* hn = hB;
    for (int i = 0; i < 6; ++i) {
        k_zero4<<<(NN * HH / 4 + 255) / 256, B, 0, stream>>>((float4*)agg, NN * HH / 4);
        k_scatter<<<(EE * 8) / 256, B, 0, stream>>>(ei, hc, agg);
        k_packA<<<(NTILE * 16 * 32) / 256, B, 0, stream>>>(agg, hc, A16);
        k_gemm<<<NTILE, B, 0, stream>>>(A16,
                                        wt16 + (size_t)i * HH * KT,
                                        bvec + (size_t)i * HH, hn);
        float* tmp = hc; hc = hn; hn = tmp;
    }

    // global mean pool + head
    k_pool_count<<<(NN + 255) / 256, B, 0, stream>>>(batch, counts);
    k_pool_sum<<<(NN * 64) / 256, B, 0, stream>>>(batch, hc, pooled);
    k_head<<<(GG * OUTD + 255) / 256, B, 0, stream>>>(pooled, counts, w_out, b_out, out);
}
